// SelfAttention_10746008174728
// MI455X (gfx1250) — matrix operand order repaired and checked by reading, not compiled, not run on hardware
//
#include <hip/hip_runtime.h>
#include <hip/hip_bf16.h>

// ---------------------------------------------------------------------------
// MI455X (gfx1250) fused self-attention: f16 WMMA, fp32 accumulate,
// flash-style online softmax, async global->LDS staging (ASYNCcnt) where
// the toolchain exposes it.
// ---------------------------------------------------------------------------

typedef _Float16 half_t;
typedef __attribute__((ext_vector_type(16))) _Float16 v16h;
typedef __attribute__((ext_vector_type(8)))  _Float16 v8h;
typedef __attribute__((ext_vector_type(8)))  float    v8f;
typedef __attribute__((ext_vector_type(4)))  int      v4i;

#if defined(__has_builtin)
#if __has_builtin(__builtin_amdgcn_global_load_async_to_lds_b128) && \
    __has_builtin(__builtin_amdgcn_s_wait_asynccnt)
#define HAS_ASYNC_LDS 1
#endif
#endif
#ifndef HAS_ASYNC_LDS
#define HAS_ASYNC_LDS 0
#endif

// 16-byte global -> LDS copy. Async (ASYNCcnt) when available.
// Builtin signature (from clang diagnostic): first param is v4i in the
// global (__device__/AS1) address space; LDS side is AS3.
__device__ __forceinline__ void cp16_async(half_t* lds, const half_t* g) {
#if HAS_ASYNC_LDS
  __builtin_amdgcn_global_load_async_to_lds_b128(
      (__attribute__((address_space(1))) v4i*)g,
      (__attribute__((address_space(3))) v4i*)lds, 0, 0);
#else
  *(v8h*)lds = *(const v8h*)g;
#endif
}

__device__ __forceinline__ void wait_async_all() {
#if HAS_ASYNC_LDS
  __builtin_amdgcn_s_wait_asynccnt(0);
#endif
}

// ---- fragment helpers (layouts per CDNA5 ISA §7.12.2, wave32) --------------

__device__ __forceinline__ v16h combine16(v8h lo, v8h hi) {
  v16h r;
#pragma unroll
  for (int i = 0; i < 8; ++i) { r[i] = lo[i]; r[8 + i] = hi[i]; }
  return r;
}

// A-matrix 16x32 f16 fragment from row-major [16][rs] halves.
// lanes 0-15: M=lane, K groups {0-7,16-23}; lanes 16-31: M=lane-16, K {8-15,24-31}
__device__ __forceinline__ v16h load_frag_a(const half_t* base, int rs) {
  const int lane = threadIdx.x & 31;
  const int m    = lane & 15;
  const int koff = (lane >> 4) * 8;
  const half_t* p = base + m * rs;
  v8h lo = *(const v8h*)(p + koff);
  v8h hi = *(const v8h*)(p + 16 + koff);
  return combine16(lo, hi);
}

// B-matrix 32x16 f16 fragment from [K=32][rs] halves (contraction-major).
// lanes 0-15: N=lane, K groups {0-7,16-23}; lanes 16-31: N=lane-16, K {8-15,24-31}
__device__ __forceinline__ v16h load_frag_b(const half_t* base, int rs) {
  const int lane = threadIdx.x & 31;
  const int koff = (lane >> 4) * 8;
  const half_t* p = base + (lane & 15);
  v16h r;
#pragma unroll
  for (int i = 0; i < 8; ++i) {
    r[i]     = p[(koff + i) * rs];
    r[8 + i] = p[(16 + koff + i) * rs];
  }
  return r;
}

__device__ __forceinline__ v8f wmma_f16(v16h a, v16h b, v8f c) {
  return __builtin_amdgcn_wmma_f32_16x16x32_f16(false, a, false, b,
                                                (short)0, c, false, false);
}

// ---- f32 -> f16 convert ----------------------------------------------------

__global__ __launch_bounds__(256) void cvt_f32_f16(const float* __restrict__ in,
                                                   half_t* __restrict__ out,
                                                   int n4) {
  int i = blockIdx.x * blockDim.x + threadIdx.x;
  if (i < n4) {
    float4 v = ((const float4*)in)[i];
    half_t* o = out + (size_t)i * 4;
    o[0] = (half_t)v.x; o[1] = (half_t)v.y;
    o[2] = (half_t)v.z; o[3] = (half_t)v.w;
  }
}

// ---- generic f16 GEMM:  C[M][N] = A[M][K] * B[K][N] + bias ----------------
// 256 threads = 8 waves (4 M x 2 N), block tile 128x128, K-step 32,
// double-buffered LDS fed by async global->LDS DMA.

template <typename OutT>
__global__ __launch_bounds__(256) void gemm_f16(const half_t* __restrict__ A,
                                                const half_t* __restrict__ B,
                                                const float* __restrict__ bias,
                                                OutT* __restrict__ C,
                                                int M, int N, int K) {
  constexpr int BM = 128, BN = 128, BK = 32;
  constexpr int ARS = BK + 8;   // 40 halves  (80 B row stride, 16B aligned)
  constexpr int BRS = BN + 8;   // 136 halves (272 B row stride, 16B aligned)
  __shared__ half_t Als[2][BM * ARS];
  __shared__ half_t Bls[2][BK * BRS];

  const int tid  = threadIdx.x;
  const int wave = tid >> 5;
  const int lane = tid & 31;
  const int wm   = wave >> 1;          // 0..3
  const int wn   = wave & 1;           // 0..1
  const int rowBase = blockIdx.y * BM;
  const int colBase = blockIdx.x * BN;

  auto stage = [&](int k0, int buf) {
    // A tile: 128x32 halves = 512 chunks of 8 halves.
#pragma unroll
    for (int i = 0; i < 2; ++i) {
      int u = tid + i * 256;
      int r = u >> 2;
      int c = (u & 3) * 8;
      cp16_async(&Als[buf][r * ARS + c],
                 A + (size_t)(rowBase + r) * K + k0 + c);
    }
    // B tile: 32x128 halves = 512 chunks of 8 halves.
#pragma unroll
    for (int i = 0; i < 2; ++i) {
      int u = tid + i * 256;
      int r = u >> 4;
      int c = (u & 15) * 8;
      cp16_async(&Bls[buf][r * BRS + c],
                 B + (size_t)(k0 + r) * N + colBase + c);
    }
  };

  v8f acc[2][4];
#pragma unroll
  for (int t = 0; t < 2; ++t)
#pragma unroll
    for (int j = 0; j < 4; ++j) acc[t][j] = (v8f)0.0f;

  const int nk = K / BK;
  stage(0, 0);

  for (int kk = 0; kk < nk; ++kk) {
    wait_async_all();     // this wave's pending tile is in LDS
    __syncthreads();      // every wave's tile is in LDS; prior buf fully read
    if (kk + 1 < nk) stage((kk + 1) * BK, (kk + 1) & 1);

    const half_t* Ab = Als[kk & 1];
    const half_t* Bb = Bls[kk & 1];
    v16h af0 = load_frag_a(Ab + (wm * 32 + 0)  * ARS, ARS);
    v16h af1 = load_frag_a(Ab + (wm * 32 + 16) * ARS, ARS);
#pragma unroll
    for (int j = 0; j < 4; ++j) {
      v16h bf = load_frag_b(Bb + wn * 64 + j * 16, BRS);
      acc[0][j] = wmma_f16(af0, bf, acc[0][j]);
      acc[1][j] = wmma_f16(af1, bf, acc[1][j]);
    }
  }

  // Epilogue: C layout — VGPR r: lanes 0-15 M=r, lanes 16-31 M=8+r; N=lane&15.
  const int rowOff = (lane >> 4) * 8;
#pragma unroll
  for (int t = 0; t < 2; ++t) {
#pragma unroll
    for (int j = 0; j < 4; ++j) {
      int col = colBase + wn * 64 + j * 16 + (lane & 15);
      float bv = bias[col];
#pragma unroll
      for (int r = 0; r < 8; ++r) {
        int row = rowBase + wm * 32 + t * 16 + rowOff + r;
        C[(size_t)row * N + col] = (OutT)(acc[t][j][r] + bv);
      }
    }
  }
}

// ---- flash attention over contiguous head slabs ---------------------------
// Q,K,V,O viewed as [128 bh][1024][96] f16 (raw-reshape => contiguous slabs).
// Block: 256 threads (8 waves); wave owns 16 q rows; block covers 128 q rows.
// blockIdx.x = bh (0..127), blockIdx.y = q tile (0..7). Keys in chunks of 32.

__global__ __launch_bounds__(256) void attn_flash(const half_t* __restrict__ Q,
                                                  const half_t* __restrict__ Km,
                                                  const half_t* __restrict__ Vm,
                                                  half_t* __restrict__ O) {
  constexpr int DK = 96;
  constexpr int KTRS = 40;    // Kt row stride (pad 32 -> 40)
  constexpr int VRS  = 104;   // V  row stride (pad 96 -> 104)
  __shared__ half_t Kt[DK * KTRS];        // transposed keys  [dim][key]
  __shared__ half_t Vl[32 * VRS];         // values           [key][dim]
  __shared__ half_t Pl[8][16 * 32];       // per-wave P scratch [m][key]

  const int tid  = threadIdx.x;
  const int wave = tid >> 5;
  const int lane = tid & 31;
  const size_t headOff = (size_t)blockIdx.x * 1024 * DK;
  const int qbase = blockIdx.y * 128 + wave * 16;

  // Q fragments (3 chunks of K=32), pre-scaled by 1/sqrt(96).
  v16h qf[3];
  {
    const int m    = lane & 15;
    const int koff = (lane >> 4) * 8;
    const half_t scale = (half_t)0.1020620726f;   // 1/sqrt(96)
    const half_t* p = Q + headOff + (size_t)(qbase + m) * DK;
#pragma unroll
    for (int c = 0; c < 3; ++c) {
      v8h lo = *(const v8h*)(p + 32 * c + koff);
      v8h hi = *(const v8h*)(p + 32 * c + 16 + koff);
#pragma unroll
      for (int i = 0; i < 8; ++i) {
        qf[c][i]     = lo[i] * scale;
        qf[c][8 + i] = hi[i] * scale;
      }
    }
  }

  v8f o_acc[6];
#pragma unroll
  for (int t = 0; t < 6; ++t) o_acc[t] = (v8f)0.0f;
  float mrow[8], lrow[8];
#pragma unroll
  for (int r = 0; r < 8; ++r) { mrow[r] = -3.0e38f; lrow[r] = 0.0f; }

  const int rowOff = (lane >> 4) * 8;
  const int n0     = lane & 15;

  for (int kt = 0; kt < 32; ++kt) {
    const half_t* Kc = Km + headOff + (size_t)(kt * 32) * DK;
    const half_t* Vc = Vm + headOff + (size_t)(kt * 32) * DK;

    // Stage V via async DMA (verbatim rows) and K via manual transpose.
    // 32 keys x 96 dims = 384 v8h chunks each.
#pragma unroll
    for (int i = 0; i < 2; ++i) {
      int idx = tid + i * 256;
      if (idx < 384) {
        int key = idx / 12;
        int c   = (idx % 12) * 8;
        cp16_async(&Vl[key * VRS + c], Vc + (size_t)key * DK + c);
        v8h dk_ = *(const v8h*)(Kc + (size_t)key * DK + c);
#pragma unroll
        for (int j = 0; j < 8; ++j) Kt[(c + j) * KTRS + key] = dk_[j];
      }
    }
    wait_async_all();
    __syncthreads();

    // S = Q @ K^T : two 16x16 tiles (keys 0-15, 16-31 of this chunk).
    v8f s0 = (v8f)0.0f, s1 = (v8f)0.0f;
#pragma unroll
    for (int c = 0; c < 3; ++c) {
      // B fragment: N = key (lane&15), elements = dims of chunk c in table order.
      const half_t* pb = &Kt[(32 * c) * KTRS];
      s0 = wmma_f16(qf[c], load_frag_b(pb, KTRS),      s0);
      s1 = wmma_f16(qf[c], load_frag_b(pb + 16, KTRS), s1);
    }

    // Online softmax update per row (row = rowOff + r in this wave's q tile).
#pragma unroll
    for (int r = 0; r < 8; ++r) {
      float mx = fmaxf(s0[r], s1[r]);
      mx = fmaxf(mx, __shfl_xor(mx, 1, 32));
      mx = fmaxf(mx, __shfl_xor(mx, 2, 32));
      mx = fmaxf(mx, __shfl_xor(mx, 4, 32));
      mx = fmaxf(mx, __shfl_xor(mx, 8, 32));
      float mnew = fmaxf(mrow[r], mx);
      float corr = __expf(mrow[r] - mnew);
      mrow[r] = mnew;
      float p0 = __expf(s0[r] - mnew);
      float p1 = __expf(s1[r] - mnew);
      s0[r] = p0; s1[r] = p1;
      float rs = p0 + p1;
      rs += __shfl_xor(rs, 1, 32);
      rs += __shfl_xor(rs, 2, 32);
      rs += __shfl_xor(rs, 4, 32);
      rs += __shfl_xor(rs, 8, 32);
      lrow[r] = lrow[r] * corr + rs;
#pragma unroll
      for (int t = 0; t < 6; ++t) o_acc[t][r] *= corr;
    }

    // P (f16) -> per-wave LDS, then reload in A-fragment layout.
    half_t* pw = Pl[wave];
#pragma unroll
    for (int r = 0; r < 8; ++r) {
      pw[(rowOff + r) * 32 + n0]      = (half_t)s0[r];
      pw[(rowOff + r) * 32 + 16 + n0] = (half_t)s1[r];
    }
    v16h pf = load_frag_a(pw, 32);   // same-wave DS ops are in-order

    // O += P @ V  (contraction over the 32 keys of this chunk).
#pragma unroll
    for (int t = 0; t < 6; ++t) {
      const half_t* pv = &Vl[16 * t];
      v16h vf = load_frag_b(pv, VRS);
      o_acc[t] = wmma_f16(pf, vf, o_acc[t]);
    }
    __syncthreads();
  }

  // Normalize and write (raw-reshape layout => contiguous slab store).
  half_t* Oh = O + headOff + (size_t)qbase * DK;
#pragma unroll
  for (int r = 0; r < 8; ++r) {
    float inv = 1.0f / lrow[r];
#pragma unroll
    for (int t = 0; t < 6; ++t)
      Oh[(size_t)(rowOff + r) * DK + 16 * t + n0] = (half_t)(o_acc[t][r] * inv);
  }
}

// ---------------------------------------------------------------------------

extern "C" void kernel_launch(void* const* d_in, const int* in_sizes, int n_in,
                              void* d_out, int out_size, void* d_ws, size_t ws_size,
                              hipStream_t stream) {
  const float* x  = (const float*)d_in[0];
  const float* Wq = (const float*)d_in[1];
  const float* bq = (const float*)d_in[2];
  const float* Wk = (const float*)d_in[3];
  const float* bk = (const float*)d_in[4];
  const float* Wv = (const float*)d_in[5];
  const float* bv = (const float*)d_in[6];
  const float* Wp = (const float*)d_in[7];
  const float* bp = (const float*)d_in[8];
  float* out = (float*)d_out;

  const int M = 16 * 1024;          // 16384 rows
  const int C = 768;
  const size_t X = (size_t)M * C;   // 12,582,912
  const size_t W = (size_t)C * C;   //    589,824

  // Workspace carve-up (halves). Total ~130.5 MB.
  half_t* ws  = (half_t*)d_ws;
  half_t* xh  = ws;
  half_t* wqh = xh  + X;
  half_t* wkh = wqh + W;
  half_t* wvh = wkh + W;
  half_t* wph = wvh + W;
  half_t* qh  = wph + W;
  half_t* kh  = qh  + X;
  half_t* vh  = kh  + X;
  half_t* ah  = vh  + X;

  // 1) fp32 -> fp16 conversions
  cvt_f32_f16<<<(int)(X / 4 / 256), 256, 0, stream>>>(x,  xh,  (int)(X / 4));
  cvt_f32_f16<<<(int)(W / 4 / 256), 256, 0, stream>>>(Wq, wqh, (int)(W / 4));
  cvt_f32_f16<<<(int)(W / 4 / 256), 256, 0, stream>>>(Wk, wkh, (int)(W / 4));
  cvt_f32_f16<<<(int)(W / 4 / 256), 256, 0, stream>>>(Wv, wvh, (int)(W / 4));
  cvt_f32_f16<<<(int)(W / 4 / 256), 256, 0, stream>>>(Wp, wph, (int)(W / 4));

  // 2) Q/K/V projections: (16384 x 768) = (16384 x 768) @ (768 x 768) + b
  dim3 gGrid(C / 128, M / 128);     // (6, 128)
  gemm_f16<half_t><<<gGrid, 256, 0, stream>>>(xh, wqh, bq, qh, M, C, C);
  gemm_f16<half_t><<<gGrid, 256, 0, stream>>>(xh, wkh, bk, kh, M, C, C);
  gemm_f16<half_t><<<gGrid, 256, 0, stream>>>(xh, wvh, bv, vh, M, C, C);

  // 3) flash attention over 128 contiguous head slabs (raw-reshape layout)
  attn_flash<<<dim3(128, 8), 256, 0, stream>>>(qh, kh, vh, ah);

  // 4) output projection (fp32 out): out = attn @ Wp + bp
  gemm_f16<float><<<gGrid, 256, 0, stream>>>(ah, wph, bp, out, M, C, C);
}
